// AttenQKlinear_63909113364812
// MI455X (gfx1250) — compile-verified, matrix operand order checked by reading, not run on hardware
//
#include <hip/hip_runtime.h>
#include <hip/hip_bf16.h>

typedef __attribute__((ext_vector_type(16))) _Float16 v16h;
typedef __attribute__((ext_vector_type(8)))  _Float16 v8h;
typedef __attribute__((ext_vector_type(8)))  float    v8f;
typedef __attribute__((ext_vector_type(4)))  int      v4i;

typedef __attribute__((address_space(1))) v4i* gptr_v4i; // global
typedef __attribute__((address_space(3))) v4i* lptr_v4i; // LDS

constexpr int kB = 8;
constexpr int kN = 2048;
constexpr int kD = 256;

constexpr int kKStride = 264; // 32x256 K-tile rows padded +8 halves (bank rotation, 16B aligned)
constexpr int kVStride = 40;  // 256x32 V-tile rows padded +8 halves

// ---------------------------------------------------------------------------
// WMMA helper: D = A(16x32 f16) * B(32x16 f16) + C(16x16 f32)
// ---------------------------------------------------------------------------
__device__ __forceinline__ v8f wmma32(v16h a, v16h b, v8f c) {
  return __builtin_amdgcn_wmma_f32_16x16x32_f16(
      /*neg_a=*/false, a, /*neg_b=*/false, b,
      /*c_mod=*/(short)0, c, /*reuse_a=*/false, /*reuse_b=*/false);
}

// 16-bit A/B fragment from row-major f16 memory (global or LDS):
// caller pre-offsets p by hi*8; halves 0..7 = p[0..7], 8..15 = p[16..23].
__device__ __forceinline__ v16h frag_from_h(const _Float16* p) {
  v8h a = *(const v8h*)p;
  v8h b = *(const v8h*)(p + 16);
  v16h r;
#pragma unroll
  for (int i = 0; i < 8; i++) { r[i] = a[i]; r[i + 8] = b[i]; }
  return r;
}

__device__ __forceinline__ v16h frag_from_f32(const float* p) {
  v16h r;
#pragma unroll
  for (int i = 0; i < 8; i++) {
    r[i]     = (_Float16)p[i];
    r[i + 8] = (_Float16)p[i + 16];
  }
  return r;
}

// ---------------------------------------------------------------------------
// Async 16-byte global->LDS copy (CDNA5 GLOBAL_LOAD_ASYNC_TO_LDS_B128,
// tracked by ASYNCcnt). Builtin sig (from hipcc diagnostic): param0 = v4i
// addrspace(1)* (global src), param1 = v4i addrspace(3)* (LDS dst), then
// imm offset, imm cpol. Fallback: synchronous copy, same barrier structure.
// ---------------------------------------------------------------------------
__device__ __forceinline__ void cp_async16(_Float16* lds_dst, const _Float16* gsrc) {
#if __has_builtin(__builtin_amdgcn_global_load_async_to_lds_b128)
  __builtin_amdgcn_global_load_async_to_lds_b128(
      (gptr_v4i)gsrc, (lptr_v4i)lds_dst, 0, 0);
#else
  *(v8h*)lds_dst = *(const v8h*)gsrc;
#endif
}

__device__ __forceinline__ void wait_async0() {
#if __has_builtin(__builtin_amdgcn_s_wait_asynccnt)
  __builtin_amdgcn_s_wait_asynccnt(0);
#else
  asm volatile("s_wait_asynccnt 0x0" ::: "memory");
#endif
}

// ---------------------------------------------------------------------------
// Kernel 1: convert Wq/Wk (f32 [256][256], torch [out,in]) to f16
// ---------------------------------------------------------------------------
__global__ void cvt_w_kernel(const float* __restrict__ Wq,
                             const float* __restrict__ Wk,
                             _Float16* __restrict__ Wq16,
                             _Float16* __restrict__ Wk16) {
  int i = blockIdx.x * blockDim.x + threadIdx.x; // 0 .. 65535
  Wq16[i] = (_Float16)Wq[i];
  Wk16[i] = (_Float16)Wk[i];
}

// ---------------------------------------------------------------------------
// Kernel 2: xT16[b][d][n] = (f16) x[b][n][d]   (V operand for PV matmul)
// ---------------------------------------------------------------------------
__global__ void xpose_kernel(const float* __restrict__ x,
                             _Float16* __restrict__ xT16) {
  int i = blockIdx.x * blockDim.x + threadIdx.x; // write-coalesced
  int n = i % kN;
  int t = i / kN;
  int d = t % kD;
  int b = t / kD;
  xT16[i] = (_Float16)x[((size_t)(b * kN + n)) * kD + d];
}

// ---------------------------------------------------------------------------
// Kernel 3: q = x @ Wq^T, k = x @ Wk^T  (f16 out, f32 accumulate via WMMA)
// ---------------------------------------------------------------------------
__global__ __launch_bounds__(128) void proj_kernel(
    const float* __restrict__ x,
    const _Float16* __restrict__ Wq16, const _Float16* __restrict__ Wk16,
    _Float16* __restrict__ q16, _Float16* __restrict__ k16) {
  int lane = threadIdx.x & 31;
  int wid  = (blockIdx.x * blockDim.x + threadIdx.x) >> 5; // 0..1023
  int b  = wid >> 7;
  int i0 = (wid & 127) << 4;
  int col = lane & 15, hi = lane >> 4;

  const float* xrow = x + ((size_t)(b * kN + i0 + col)) * kD + hi * 8;
  v16h af[8];
#pragma unroll
  for (int kc = 0; kc < 8; kc++) af[kc] = frag_from_f32(xrow + kc * 32);

#pragma unroll 1
  for (int m = 0; m < 2; m++) {
    const _Float16* W   = m ? Wk16 : Wq16;
    _Float16*       dst = m ? k16  : q16;
    const _Float16* wrow = W + (size_t)col * kD + hi * 8;
#pragma unroll 1
    for (int nt = 0; nt < 16; nt++) {
      v8f c = {};
      const _Float16* wr = wrow + (size_t)(nt * 16) * kD;
#pragma unroll
      for (int kc = 0; kc < 8; kc++) {
        c = wmma32(af[kc], frag_from_h(wr + kc * 32), c);
      }
#pragma unroll
      for (int r = 0; r < 8; r++) {
        dst[((size_t)(b * kN + i0 + r + 8 * hi)) * kD + nt * 16 + col] =
            (_Float16)c[r];
      }
    }
  }
}

// ---------------------------------------------------------------------------
// Kernel 4: flash attention, 8 waves/block. Each wave owns 16 query rows;
// the block cooperatively double-buffers the shared K-tile (32x256) and
// V-tile (256x32) through LDS with async copies, cutting L2 re-read traffic
// 8x vs per-wave streaming. Online softmax per wave.
// ---------------------------------------------------------------------------
__global__ __launch_bounds__(256) void fa_kernel(
    const _Float16* __restrict__ q16, const _Float16* __restrict__ k16,
    const _Float16* __restrict__ xT16, float* __restrict__ out) {
  __shared__ __attribute__((aligned(16))) _Float16 kT[2][32 * kKStride];
  __shared__ __attribute__((aligned(16))) _Float16 vT[2][kD * kVStride];
  __shared__ float sm[8][16 * 32 + 16]; // per-wave score tile + row stats

  int tid   = threadIdx.x;
  int lane  = tid & 31;
  int wslot = tid >> 5;
  int wid   = (blockIdx.x * blockDim.x + tid) >> 5;
  int b  = wid >> 7;
  int i0 = (wid & 127) << 4;
  int col = lane & 15, hi = lane >> 4;
  float* S    = &sm[wslot][0];
  float* stat = &sm[wslot][512];
  const float scale = 0.0625f; // d^-0.5

  const _Float16* kb0 = k16  + (size_t)b * kN * kD;
  const _Float16* vb0 = xT16 + (size_t)b * kD * kN;

  // cooperative staging of one (K,V) column tile j0s..j0s+31 into buffer `buf`
  auto stage = [&](int buf, int j0s) {
#pragma unroll
    for (int i = 0; i < 4; i++) {         // K tile: 1024 x 16B chunks
      int c = tid + i * 256;
      int r = c >> 5, cc = (c & 31) * 8;
      cp_async16(&kT[buf][r * kKStride + cc],
                 kb0 + (size_t)(j0s + r) * kD + cc);
    }
#pragma unroll
    for (int i = 0; i < 4; i++) {         // V tile: 1024 x 16B chunks
      int c = tid + i * 256;
      int d = c >> 2, jj = (c & 3) * 8;
      cp_async16(&vT[buf][d * kVStride + jj],
                 vb0 + (size_t)d * kN + j0s + jj);
    }
  };

  // Q fragments held in registers for the whole sweep (64 VGPRs)
  const _Float16* qrow = q16 + ((size_t)(b * kN + i0 + col)) * kD + hi * 8;
  v16h qf[8];
#pragma unroll
  for (int kc = 0; kc < 8; kc++) qf[kc] = frag_from_h(qrow + kc * 32);

  v8f o[16];
#pragma unroll
  for (int t = 0; t < 16; t++) o[t] = (v8f){};
  float mrow = -1e30f, lrow = 0.0f; // valid in lanes 0..15

  stage(0, 0);
  wait_async0();
  __syncthreads();

#pragma unroll 1
  for (int it = 0; it < kN / 32; ++it) {
    int cur = it & 1;
    if (it + 1 < kN / 32) stage(cur ^ 1, (it + 1) * 32); // prefetch next tile

    const _Float16* kt = &kT[cur][0];
    const _Float16* vt = &vT[cur][0];

    // ---- S tile: Q(16x256) x K^T(256x32) via 16 WMMAs, B frags from LDS ----
    v8f s0 = {}, s1 = {};
    const _Float16* kr0 = kt + col * kKStride + hi * 8;
    const _Float16* kr1 = kt + (col + 16) * kKStride + hi * 8;
#pragma unroll
    for (int kc = 0; kc < 8; kc++) {
      s0 = wmma32(qf[kc], frag_from_h(kr0 + kc * 32), s0);
      s1 = wmma32(qf[kc], frag_from_h(kr1 + kc * 32), s1);
    }
    // scatter C-layout scores into per-wave LDS (scaled)
#pragma unroll
    for (int r = 0; r < 8; r++) {
      S[(r + 8 * hi) * 32 + col]      = s0[r] * scale;
      S[(r + 8 * hi) * 32 + col + 16] = s1[r] * scale;
    }
    asm volatile("s_wait_dscnt 0x0" ::: "memory"); // wave-internal LDS RAW

    // ---- online softmax: lane l<16 owns query row l ----
    if (lane < 16) {
      float* Srow = S + lane * 32;
      float sv[32];
      float tm = mrow;
#pragma unroll
      for (int j = 0; j < 32; j++) { sv[j] = Srow[j]; tm = fmaxf(tm, sv[j]); }
      float fac = __expf(mrow - tm);
      float ssum = 0.0f;
#pragma unroll
      for (int j = 0; j < 32; j++) {
        float p = __expf(sv[j] - tm);
        Srow[j] = p;
        ssum += p;
      }
      lrow = lrow * fac + ssum;
      mrow = tm;
      stat[lane] = fac;
    }
    asm volatile("s_wait_dscnt 0x0" ::: "memory");

    // ---- rescale O accumulators by per-row factor ----
    float ffac[8];
#pragma unroll
    for (int r = 0; r < 8; r++) ffac[r] = stat[8 * hi + r];
#pragma unroll
    for (int t = 0; t < 16; t++) {
#pragma unroll
      for (int r = 0; r < 8; r++) o[t][r] *= ffac[r];
    }

    // ---- P fragment in A-layout from per-wave LDS ----
    const float* Prow = S + col * 32 + hi * 8;
    v16h pf;
#pragma unroll
    for (int h = 0; h < 8; h++) {
      pf[h]     = (_Float16)Prow[h];
      pf[h + 8] = (_Float16)Prow[h + 16];
    }

    // ---- O += P(16x32) @ V(32x256), B frags from LDS V-tile ----
#pragma unroll
    for (int t = 0; t < 16; t++) {
      o[t] = wmma32(pf, frag_from_h(vt + (t * 16 + col) * kVStride + hi * 8),
                    o[t]);
    }

    // own async staging done + all waves finished reading `cur` before the
    // next tile becomes visible / this buffer is overwritten next round
    wait_async0();
    __syncthreads();
  }

  // ---- epilogue: divide by row sums, store f32 ----
  if (lane < 16) stat[lane] = 1.0f / lrow;
  asm volatile("s_wait_dscnt 0x0" ::: "memory");
  float rinv[8];
#pragma unroll
  for (int r = 0; r < 8; r++) rinv[r] = stat[8 * hi + r];
#pragma unroll
  for (int t = 0; t < 16; t++) {
#pragma unroll
    for (int r = 0; r < 8; r++) {
      out[((size_t)(b * kN + i0 + r + 8 * hi)) * kD + t * 16 + col] =
          o[t][r] * rinv[r];
    }
  }
}

// ---------------------------------------------------------------------------
// Launch: inputs: x[8,2048,256] f32, Wq[256,256], Wk[256,256].
// Output: [8,2048,256] f32. Workspace: q16 | k16 | xT16 | Wq16 | Wk16 (~25MB)
// ---------------------------------------------------------------------------
extern "C" void kernel_launch(void* const* d_in, const int* in_sizes, int n_in,
                              void* d_out, int out_size, void* d_ws, size_t ws_size,
                              hipStream_t stream) {
  (void)in_sizes; (void)n_in; (void)out_size; (void)ws_size;
  const float* x  = (const float*)d_in[0];
  const float* Wq = (const float*)d_in[1];
  const float* Wk = (const float*)d_in[2];
  float* out = (float*)d_out;

  const size_t nqk = (size_t)kB * kN * kD; // 4,194,304 elems
  _Float16* q16  = (_Float16*)d_ws;
  _Float16* k16  = q16  + nqk;
  _Float16* xT16 = k16  + nqk;
  _Float16* Wq16 = xT16 + nqk;
  _Float16* Wk16 = Wq16 + (size_t)kD * kD;

  cvt_w_kernel<<<(kD * kD) / 256, 256, 0, stream>>>(Wq, Wk, Wq16, Wk16);
  xpose_kernel<<<(kB * kD * kN) / 256, 256, 0, stream>>>(x, xT16);
  proj_kernel<<<256, 128, 0, stream>>>(x, Wq16, Wk16, q16, k16);
  // 1024 waves = 128 blocks x 8 waves; each block shares K/V tiles via LDS
  fa_kernel<<<128, 256, 0, stream>>>(q16, k16, xT16, out);
}
